// MyNet_67929202753936
// MI455X (gfx1250) — matrix operand order repaired and checked by reading, not compiled, not run on hardware
//
#include <hip/hip_runtime.h>
#include <hip/hip_bf16.h>

// LeNet-5 forward, fully fused, batch-as-M WMMA mapping for gfx1250 (MI455X).
// Each wave processes 16 images end-to-end in its private LDS slice.
// All matmul work (conv1/conv2 as implicit GEMM, FC1-3) runs on
// v_wmma_f32_16x16x32_f16 with f32 accumulation.
//
// Rev 3: biases are hoisted out of the pooled-position loops as clamped
// unconditional loads (the previous version re-loaded b1[lm] under a
// divergent branch with an s_wait_loadcnt 0 inside the 144-iteration
// conv1 loop). Gathers read zero-padded LDS unconditionally; tap
// decomposition folds at compile time; B / FC-A fragments use paired-u32
// layouts (ds_load_b32).

typedef __attribute__((ext_vector_type(16))) _Float16     v16h;
typedef __attribute__((ext_vector_type(8)))  float        v8f;
typedef __attribute__((ext_vector_type(8)))  unsigned int v8u;

// ---- geometry -----------------------------------------------------------
#define IMG_PIX   784                   // 28x28
#define XS_ROWS   840                   // 784 valid + pad rows (k<32 overreach)
#define H1_ROWS   1544                  // 1440 valid + pad rows (k<256 overreach)
#define XS_HALVES (XS_ROWS * 16)        // 13440
#define H1_HALVES (H1_ROWS * 16)        // 24704
#define WAVE_HALVES (XS_HALVES + H1_HALVES)   // 38144 halves = 76288 B
#define W1S_HALVES  (32 * 16)           // paired u32 [16 kp][16 n]
#define W2S_HALVES  (256 * 32)          // paired u32 [128 kp][32 n]
#define SMEM_HALVES (W1S_HALVES + W2S_HALVES + 4 * WAVE_HALVES)
#define SMEM_BYTES  (SMEM_HALVES * 2)   // 322560 <= 327680 (320 KB WGP)

__device__ __forceinline__ float sigmoidf(float v) {
    return 1.0f / (1.0f + __expf(-v));
}

// compile-time tap -> LDS row decomposition (folded after unrolling)
__host__ __device__ constexpr int c1row(int k) {            // conv1: 28-wide img
    return (k / 5) * 28 + (k % 5);
}
__host__ __device__ constexpr int c2row(int k) {            // conv2: 10ch 12x12
    return (k / 25) * 144 + ((k % 25) / 5) * 12 + (k % 5);
}

__device__ __forceinline__ v8f wmma(v16h a, v16h b, v8f c) {
    return __builtin_amdgcn_wmma_f32_16x16x32_f16(
        false, a, false, b, (short)0, c, false, false);
}

// A fragment from paired-u32 activation [kp][16 images]: 8x ds_load_b32.
// Slot i holds k = kb*32 + ((i<4)?2i:16+2(i-4)) + lh*8 (even) and k+1.
__device__ __forceinline__ v16h ldsAp(const unsigned* base, int lm, int lh,
                                      int kb) {
    union { v8u u; v16h h; } r;
#pragma unroll
    for (int i = 0; i < 8; ++i) {
        int kp = kb * 16 + lh * 4 + ((i < 4) ? i : i + 4);
        r.u[i] = base[kp * 16 + lm];
    }
    return r.h;
}

// B fragment from paired-u32 weight tile [kp][ncols]: 8x ds_load_b32.
// Slot i holds k = kb*32 + ((i<4)?2i:16+2(i-4)) + lh*8 (even) and k+1.
__device__ __forceinline__ v16h ldsBp(const unsigned* base, int ncols, int col,
                                      int lh, int kb) {
    union { v8u u; v16h h; } r;
#pragma unroll
    for (int i = 0; i < 8; ++i) {
        int kp = kb * 16 + lh * 4 + ((i < 4) ? i : i + 4);
        r.u[i] = base[kp * ncols + col];
    }
    return r.h;
}

// B fragment from a global f32 weight [K][N] (L2-resident). Unconditional
// clamped loads + value selects: no exec-mask branches. kb must be a
// compile-time constant at call sites so bounds checks fold.
template <int K, int N>
__device__ __forceinline__ v16h glbB(const float* __restrict__ W, int col,
                                     int lh, int kb) {
    v16h b;
    const bool cok = (col < N);
    const int  cc  = cok ? col : (N - 1);
#pragma unroll
    for (int i = 0; i < 8; ++i) {
        int k  = kb * 32 + lh * 8 + ((i < 4) ? 2 * i : 16 + 2 * (i - 4));
        int k0 = (k < K) ? k : (K - 1);
        int k1 = (k + 1 < K) ? (k + 1) : (K - 1);
        float v0 = W[k0 * N + cc];
        float v1 = W[k1 * N + cc];
        b[2 * i]     = (cok && k     < K) ? (_Float16)v0 : (_Float16)0.0f;
        b[2 * i + 1] = (cok && k + 1 < K) ? (_Float16)v1 : (_Float16)0.0f;
    }
    return b;
}

__global__ __launch_bounds__(128, 1) void lenet5_fused(
    const float* __restrict__ x,  const float* __restrict__ W1,
    const float* __restrict__ b1, const float* __restrict__ W2,
    const float* __restrict__ b2, const float* __restrict__ L1,
    const float* __restrict__ Lb1, const float* __restrict__ L2,
    const float* __restrict__ Lb2, const float* __restrict__ L3,
    const float* __restrict__ Lb3, float* __restrict__ out) {
    extern __shared__ _Float16 smem[];

    const int tid  = threadIdx.x;
    const int wave = tid >> 5;
    const int lane = tid & 31;
    const int lm   = lane & 15;   // A row (image) / B,D column
    const int lh   = lane >> 4;   // lane half selects K (A/B) and M (D) range

    _Float16* w1h = smem;                     // paired [16 kp][16 n]
    _Float16* w2h = w1h + W1S_HALVES;         // paired [128 kp][32 n]
    _Float16* wb  = w2h + W2S_HALVES + wave * WAVE_HALVES;
    _Float16* xs  = wb;                       // [840 rows][16], rows>=784 zero
    _Float16* h1s = wb + XS_HALVES;           // [1544 rows][16], rows>=1440 zero
    // aliases inside xs region once conv1 has consumed it:
    _Float16* h2h = xs;                       // paired, 320 feats -> 5120 halves
    _Float16* f1h = xs + 5120;                // paired, 128 feats -> 2048 halves
    _Float16* f2h = xs + 7168;                // paired, 96 feats  -> 1536 halves

    // hoisted, clamped bias loads (one wait total, no exec-mask branching)
    const float bb1  = b1[lm < 10 ? lm : 0];
    const float bb2a = b2[lm];                       // lm < 16 < 20 always ok
    const float bb2b = b2[lm < 4 ? 16 + lm : 16];

    // ---- phase 0: conv weights -> padded f16 paired B-layout in LDS -----
    for (int idx = tid; idx < W1S_HALVES; idx += blockDim.x) {
        int par = idx & 1, pair = idx >> 1;
        int kp = pair >> 4, col = pair & 15, k = 2 * kp + par;
        w1h[idx] = (_Float16)((k < 25 && col < 10) ? W1[col * 25 + k] : 0.0f);
    }
    for (int idx = tid; idx < W2S_HALVES; idx += blockDim.x) {
        int par = idx & 1, pair = idx >> 1;
        int kp = pair >> 5, col = pair & 31, k = 2 * kp + par;
        w2h[idx] = (_Float16)((k < 250 && col < 20) ? W2[col * 250 + k] : 0.0f);
    }

    const int g0 = blockIdx.x * 64 + wave * 16;   // first image of this wave

    // ---- phase 1: stage 16 images transposed [pixel][image]; zero pads --
    for (int m = 0; m < 16; ++m) {
        const float4* xi = (const float4*)(x + (size_t)(g0 + m) * IMG_PIX);
        for (int q = lane; q < IMG_PIX / 4; q += 32) {
            float4 v = xi[q];
            int r = q * 4;
            xs[(r + 0) * 16 + m] = (_Float16)v.x;
            xs[(r + 1) * 16 + m] = (_Float16)v.y;
            xs[(r + 2) * 16 + m] = (_Float16)v.z;
            xs[(r + 3) * 16 + m] = (_Float16)v.w;
        }
    }
    for (int idx = lane; idx < (XS_ROWS - IMG_PIX) * 16; idx += 32)
        xs[IMG_PIX * 16 + idx] = (_Float16)0.0f;
    for (int idx = lane; idx < (H1_ROWS - 1440) * 16; idx += 32)
        h1s[1440 * 16 + idx] = (_Float16)0.0f;
    __syncthreads();

    // ---- phase 2: conv1 (5x5, 1->10) + 2x2 avgpool + sigmoid ------------
    {
        v16h bf = ldsBp((const unsigned*)w1h, 16, lm, lh, 0);
        int offs[16];                       // tap offsets, constants + lm
#pragma unroll
        for (int i = 0; i < 8; ++i)
#pragma unroll
            for (int j = 0; j < 2; ++j) {
                const int k0 = ((i < 4) ? 2 * i : 16 + 2 * (i - 4)) + j;
                int row = lh ? c1row(k0 + 8) : c1row(k0);   // both constexpr
                offs[2 * i + j] = row * 16 + lm;
            }
        for (int pr = 0; pr < 12; ++pr)
            for (int pc = 0; pc < 12; ++pc) {
                v8f acc = {};
#pragma unroll
                for (int s = 0; s < 4; ++s) {
                    int r = 2 * pr + (s >> 1), c = 2 * pc + (s & 1);
                    int pb = (r * 28 + c) * 16;
                    v16h a;
#pragma unroll
                    for (int e = 0; e < 16; ++e)
                        a[e] = xs[offs[e] + pb];            // unconditional
                    acc = wmma(a, bf, acc);
                }
                if (lm < 10) {                  // D column = output channel
#pragma unroll
                    for (int i = 0; i < 8; ++i) {
                        int mi = lh * 8 + i;    // D row = image
                        h1s[(lm * 144 + pr * 12 + pc) * 16 + mi] =
                            (_Float16)sigmoidf(acc[i] * 0.25f + bb1);
                    }
                }
            }
    }
    __syncthreads();

    // zero FC pad rows (region aliases xs, free now that conv1 is done)
    for (int idx = lane; idx < 128; idx += 32)   // f1 feats 120..127
        f1h[1920 + idx] = (_Float16)0.0f;
    for (int idx = lane; idx < 192; idx += 32)   // f2 feats 84..95
        f2h[1344 + idx] = (_Float16)0.0f;

    // ---- phase 3: conv2 (5x5, 10->20) + 2x2 avgpool + sigmoid -----------
    // K = 250 (pad 256, 8 k-steps), N = 20 (2 n-tiles sharing A fragments).
    for (int pp = 0; pp < 16; ++pp) {
        int pr = pp >> 2, pc = pp & 3;
        v8f acc0 = {}, acc1 = {};
#pragma unroll
        for (int kb = 0; kb < 8; ++kb) {
            v16h bf0 = ldsBp((const unsigned*)w2h, 32, lm,      lh, kb);
            v16h bf1 = ldsBp((const unsigned*)w2h, 32, lm + 16, lh, kb);
            int offs[16];
#pragma unroll
            for (int i = 0; i < 8; ++i)
#pragma unroll
                for (int j = 0; j < 2; ++j) {
                    const int k0 =
                        kb * 32 + ((i < 4) ? 2 * i : 16 + 2 * (i - 4)) + j;
                    int row = lh ? c2row(k0 + 8) : c2row(k0);  // constexpr
                    offs[2 * i + j] = row * 16 + lm;
                }
#pragma unroll
            for (int s = 0; s < 4; ++s) {
                int r = 2 * pr + (s >> 1), c = 2 * pc + (s & 1);
                int pb = (r * 12 + c) * 16;
                v16h a;
#pragma unroll
                for (int e = 0; e < 16; ++e)
                    a[e] = h1s[offs[e] + pb];                // unconditional
                acc0 = wmma(a, bf0, acc0);
                acc1 = wmma(a, bf1, acc1);
            }
        }
        // h2 feature index = ch*16 + pp, stored in paired layout
        {
            int kf = lm * 16 + pp;
#pragma unroll
            for (int i = 0; i < 8; ++i)
                h2h[(kf >> 1) * 32 + (lh * 8 + i) * 2 + (kf & 1)] =
                    (_Float16)sigmoidf(acc0[i] * 0.25f + bb2a);
        }
        if (lm < 4) {
            int kf = (16 + lm) * 16 + pp;
#pragma unroll
            for (int i = 0; i < 8; ++i)
                h2h[(kf >> 1) * 32 + (lh * 8 + i) * 2 + (kf & 1)] =
                    (_Float16)sigmoidf(acc1[i] * 0.25f + bb2b);
        }
    }
    __syncthreads();

    // ---- phase 4: FC1  [16,320] @ [320,120] + sigmoid -------------------
    for (int g = 0; g < 2; ++g) {
        v8f acc[4] = {};
#pragma unroll
        for (int kb = 0; kb < 10; ++kb) {
            v16h a = ldsAp((const unsigned*)h2h, lm, lh, kb);
#pragma unroll
            for (int q = 0; q < 4; ++q) {
                v16h b = glbB<320, 120>(L1, (g * 4 + q) * 16 + lm, lh, kb);
                acc[q] = wmma(a, b, acc[q]);
            }
        }
#pragma unroll
        for (int q = 0; q < 4; ++q) {
            int n = (g * 4 + q) * 16 + lm;
            float bb = Lb1[n < 120 ? n : 0];   // clamped, unconditional
            if (n < 120) {
#pragma unroll
                for (int i = 0; i < 8; ++i)
                    f1h[(n >> 1) * 32 + (lh * 8 + i) * 2 + (n & 1)] =
                        (_Float16)sigmoidf(acc[q][i] + bb);
            }
        }
    }
    __syncthreads();

    // ---- phase 5: FC2  [16,120] @ [120,84] + sigmoid --------------------
    for (int nt = 0; nt < 6; ++nt) {
        v8f acc = {};
#pragma unroll
        for (int kb = 0; kb < 4; ++kb) {
            v16h a = ldsAp((const unsigned*)f1h, lm, lh, kb);
            v16h b = glbB<120, 84>(L2, nt * 16 + lm, lh, kb);
            acc = wmma(a, b, acc);
        }
        int n = nt * 16 + lm;
        float bb = Lb2[n < 84 ? n : 0];        // clamped, unconditional
        if (n < 84) {
#pragma unroll
            for (int i = 0; i < 8; ++i)
                f2h[(n >> 1) * 32 + (lh * 8 + i) * 2 + (n & 1)] =
                    (_Float16)sigmoidf(acc[i] + bb);
        }
    }
    __syncthreads();

    // ---- phase 6: FC3  [16,84] @ [84,10] (no activation) ----------------
    {
        v8f acc = {};
#pragma unroll
        for (int kb = 0; kb < 3; ++kb) {
            v16h a = ldsAp((const unsigned*)f2h, lm, lh, kb);
            v16h b = glbB<84, 10>(L3, lm, lh, kb);
            acc = wmma(a, b, acc);
        }
        float bb = Lb3[lm < 10 ? lm : 0];      // clamped, unconditional
        if (lm < 10) {
#pragma unroll
            for (int i = 0; i < 8; ++i)
                out[(size_t)(g0 + lh * 8 + i) * 10 + lm] = acc[i] + bb;
        }
    }
}

extern "C" void kernel_launch(void* const* d_in, const int* in_sizes, int n_in,
                              void* d_out, int out_size, void* d_ws, size_t ws_size,
                              hipStream_t stream) {
    (void)n_in; (void)out_size; (void)d_ws; (void)ws_size;
    const float* x   = (const float*)d_in[0];
    const float* W1  = (const float*)d_in[1];
    const float* b1  = (const float*)d_in[2];
    const float* W2  = (const float*)d_in[3];
    const float* b2  = (const float*)d_in[4];
    const float* L1  = (const float*)d_in[5];
    const float* Lb1 = (const float*)d_in[6];
    const float* L2  = (const float*)d_in[7];
    const float* Lb2 = (const float*)d_in[8];
    const float* L3  = (const float*)d_in[9];
    const float* Lb3 = (const float*)d_in[10];
    float* out = (float*)d_out;

    const int nimg   = in_sizes[0] / IMG_PIX;   // 8192
    const int blocks = nimg / 64;               // 16 img/wave * 4 waves/block

    (void)hipFuncSetAttribute((const void*)lenet5_fused,
                              hipFuncAttributeMaxDynamicSharedMemorySize,
                              (int)SMEM_BYTES);

    lenet5_fused<<<blocks, 128, SMEM_BYTES, stream>>>(
        x, W1, b1, W2, b2, L1, Lb1, L2, Lb2, L3, Lb3, out);
}
